// ResidualVectorQuantizer_33526514712761
// MI455X (gfx1250) — matrix operand order repaired and checked by reading, not compiled, not run on hardware
//
#include <hip/hip_runtime.h>
#include <hip/hip_bf16.h>
#include <stdint.h>

// ---- problem constants (match reference) ----
#define B_   16
#define T_   4096
#define C_   256
#define V_   1024
#define K_   4
#define NTOK (B_ * T_)          // 65536 tokens

typedef __attribute__((ext_vector_type(16))) __bf16 v16bf;
typedef __attribute__((ext_vector_type(8)))  float  v8f;

__device__ __forceinline__ unsigned short f2bf(float f) {
  unsigned u = __builtin_bit_cast(unsigned, f);
  u += 0x7FFFu + ((u >> 16) & 1u);   // round-to-nearest-even
  return (unsigned short)(u >> 16);
}

// ---- kernel 1: codebook -> bf16, ||c_v||^2, zero loss accumulator ----
__global__ void rvq_prep_cb(const float* __restrict__ cb,
                            unsigned short* __restrict__ cbh,
                            float* __restrict__ c2,
                            float* __restrict__ lossAcc) {
  __shared__ float sm[256];
  const int row = blockIdx.x;          // 0 .. K_*V_-1
  const int c   = threadIdx.x;
  float x = cb[(size_t)row * C_ + c];
  cbh[(size_t)row * C_ + c] = f2bf(x);
  sm[c] = x * x;
  __syncthreads();
  for (int s = 128; s > 0; s >>= 1) { if (c < s) sm[c] += sm[c + s]; __syncthreads(); }
  if (c == 0) c2[row] = sm[0];
  if (row == 0 && c == 0) *lossAcc = 0.0f;
}

// ---- kernel 2: transpose z [B,C,T] -> residual R/Rh token-major [B*T, C] ----
__global__ void rvq_init(const float* __restrict__ z,
                         float* __restrict__ R,
                         unsigned short* __restrict__ Rh) {
  size_t o = (size_t)blockIdx.x * blockDim.x + threadIdx.x;  // z-layout index
  int tt = (int)(o % T_);
  int c  = (int)((o / T_) % C_);
  int b  = (int)(o / ((size_t)C_ * T_));
  float x = z[o];
  size_t ro = ((size_t)(b * T_ + tt)) * C_ + c;
  R[ro]  = x;
  Rh[ro] = f2bf(x);
}

// ---- kernel 3: WMMA distance GEMM + argmin over 1024 codewords ----
// block = 8 waves; each wave owns 32 tokens (2 M-tiles, A resident in VGPRs).
// B tiles (16 codewords x 256 dims bf16 = 8KB) staged block-wide into LDS,
// double-buffered; every LDS B-fragment feeds two WMMA chains.
union Frag { v16bf v; unsigned u[8]; };

__global__ void __launch_bounds__(256)
rvq_argmin(const unsigned short* __restrict__ Rh,
           const unsigned short* __restrict__ cbh,   // level base [V_, C_] bf16
           const float* __restrict__ c2,             // level base [V_]
           int* __restrict__ idxbuf,
           float* __restrict__ outIdx,               // d_out indices region
           int lvl) {
  constexpr int RS = 264;                 // row stride in bf16 (528B: 4-bank skew)
  __shared__ __align__(16) unsigned short Bt[2][16 * RS];   // 2 x 8.25KB

  const int tid  = threadIdx.x;
  const int lane = tid & 31;
  const int wave = tid >> 5;
  const int t0   = (blockIdx.x * 8 + wave) * 32;   // 32 tokens per wave
  const int n    = lane & 15;                      // M idx for A, N idx for B
  const int half = lane >> 4;

  // ---- resident A fragments for 2 M-tiles (ISA 16-bit A layout) ----
  Frag a0[8], a1[8];
  const unsigned short* ar0 = Rh + (size_t)(t0 + n) * C_;
  const unsigned short* ar1 = Rh + (size_t)(t0 + 16 + n) * C_;
#pragma unroll
  for (int kc = 0; kc < 8; ++kc) {
#pragma unroll
    for (int j = 0; j < 8; ++j) {
      int k = kc * 32 + ((j < 4) ? 0 : 16) + half * 8 + 2 * (j & 3);
      a0[kc].u[j] = *(const unsigned*)(ar0 + k);
      a1[kc].u[j] = *(const unsigned*)(ar1 + k);
    }
  }

  float minv0[8], minv1[8];
  int   mini0[8], mini1[8];
#pragma unroll
  for (int e = 0; e < 8; ++e) {
    minv0[e] = minv1[e] = 3.4e38f;
    mini0[e] = mini1[e] = 0;
  }

  // ---- staging: each thread copies 32B of the 8KB tile ----
  const int srow = tid >> 4;              // 0..15 codeword row in tile
  const int scol = (tid & 15) * 16;       // bf16 column offset
  const unsigned short* gsrc = cbh + (size_t)srow * C_ + scol;
  unsigned short* lbase = &Bt[0][0] + srow * RS + scol;

  uint4 p0 = *(const uint4*)(gsrc);
  uint4 p1 = *(const uint4*)(gsrc + 8);
  *(uint4*)(lbase)     = p0;
  *(uint4*)(lbase + 8) = p1;
  __syncthreads();

  for (int nt = 0; nt < 64; ++nt) {
    // prefetch next tile into registers (overlaps with WMMA below)
    if (nt + 1 < 64) {
      const unsigned short* g = gsrc + (size_t)(nt + 1) * 16 * C_;
      p0 = *(const uint4*)(g);
      p1 = *(const uint4*)(g + 8);
    }

    const unsigned short* bbuf = &Bt[nt & 1][0] + n * RS + half * 8;
    v8f acc0 = {0.f, 0.f, 0.f, 0.f, 0.f, 0.f, 0.f, 0.f};
    v8f acc1 = {0.f, 0.f, 0.f, 0.f, 0.f, 0.f, 0.f, 0.f};
#pragma unroll
    for (int kc = 0; kc < 8; ++kc) {
      Frag bf;                             // ISA 16-bit B layout: k 8h..8h+7, 16+8h..16+8h+7
      *(uint4*)&bf.u[0] = *(const uint4*)(bbuf + kc * 32);
      *(uint4*)&bf.u[4] = *(const uint4*)(bbuf + kc * 32 + 16);
      acc0 = __builtin_amdgcn_wmma_f32_16x16x32_bf16(false, a0[kc].v, false, bf.v,
                                                     (short)0, acc0, false, false);
      acc1 = __builtin_amdgcn_wmma_f32_16x16x32_bf16(false, a1[kc].v, false, bf.v,
                                                     (short)0, acc1, false, false);
    }

    const int   v  = nt * 16 + n;
    const float cc = c2[v];
#pragma unroll
    for (int e = 0; e < 8; ++e) {
      float s0 = cc - 2.0f * acc0[e];      // ||c||^2 - 2 r.c
      float s1 = cc - 2.0f * acc1[e];
      if (s0 < minv0[e]) { minv0[e] = s0; mini0[e] = v; }
      if (s1 < minv1[e]) { minv1[e] = s1; mini1[e] = v; }
    }

    if (nt + 1 < 64) {
      __syncthreads();                     // all waves done reading buf[nt&1]^1
      unsigned short* d = lbase + ((nt + 1) & 1) * (16 * RS);
      *(uint4*)(d)     = p0;
      *(uint4*)(d + 8) = p1;
      __syncthreads();
    }
  }

  // ---- reduce (min, argmin) across the 16 lanes of each half ----
#pragma unroll
  for (int e = 0; e < 8; ++e) {
#pragma unroll
    for (int off = 1; off < 16; off <<= 1) {
      float ov = __shfl_xor(minv0[e], off, 32);
      int   oi = __shfl_xor(mini0[e], off, 32);
      if (ov < minv0[e] || (ov == minv0[e] && oi < mini0[e])) { minv0[e] = ov; mini0[e] = oi; }
      ov = __shfl_xor(minv1[e], off, 32);
      oi = __shfl_xor(mini1[e], off, 32);
      if (ov < minv1[e] || (ov == minv1[e] && oi < mini1[e])) { minv1[e] = ov; mini1[e] = oi; }
    }
  }

  if (n == 0) {
#pragma unroll
    for (int e = 0; e < 8; ++e) {
      int tok0 = t0 + e + 8 * half;        // D row m = vgpr + 8*half
      int tok1 = tok0 + 16;
      idxbuf[tok0] = mini0[e];
      idxbuf[tok1] = mini1[e];
      int b0 = tok0 >> 12, tt0 = tok0 & (T_ - 1);
      int b1 = tok1 >> 12, tt1 = tok1 & (T_ - 1);
      outIdx[((size_t)(b0 * K_ + lvl)) * T_ + tt0] = (float)mini0[e];
      outIdx[((size_t)(b1 * K_ + lvl)) * T_ + tt1] = (float)mini1[e];
    }
  }
}

// ---- kernel 4: residual update + commitment-loss partial sums ----
__global__ void rvq_update(float* __restrict__ R,
                           unsigned short* __restrict__ Rh,
                           const float* __restrict__ cb,    // level base [V_,C_] fp32
                           const int* __restrict__ idxbuf,
                           float* __restrict__ lossAcc) {
  __shared__ float sm[256];
  const int t = blockIdx.x;
  const int c = threadIdx.x;
  const int idx = idxbuf[t];
  size_t ro = (size_t)t * C_ + c;
  float q  = cb[(size_t)idx * C_ + c];
  float rn = R[ro] - q;                  // new residual; loss term = mean(rn^2)
  R[ro]  = rn;
  Rh[ro] = f2bf(rn);
  sm[c] = rn * rn;
  __syncthreads();
  for (int s = 128; s > 0; s >>= 1) { if (c < s) sm[c] += sm[c + s]; __syncthreads(); }
  if (c == 0) atomicAdd(lossAcc, sm[0]);
}

// ---- kernel 5: z_q = z - r_final (back to [B,C,T]) + final loss ----
__global__ void rvq_final(const float* __restrict__ z,
                          const float* __restrict__ R,
                          float* __restrict__ out,
                          const float* __restrict__ lossAcc) {
  size_t o = (size_t)blockIdx.x * blockDim.x + threadIdx.x;  // out/z layout
  int tt = (int)(o % T_);
  int c  = (int)((o / T_) % C_);
  int b  = (int)(o / ((size_t)C_ * T_));
  size_t ro = ((size_t)(b * T_ + tt)) * C_ + c;
  out[o] = z[o] - R[ro];
  if (o == 0) {
    const float scale = 1.0f / ((float)K_ * (float)((size_t)B_ * T_ * C_));
    out[(size_t)B_ * C_ * T_ + (size_t)B_ * K_ * T_] = *lossAcc * scale;
  }
}

extern "C" void kernel_launch(void* const* d_in, const int* in_sizes, int n_in,
                              void* d_out, int out_size, void* d_ws, size_t ws_size,
                              hipStream_t stream) {
  const float* z  = (const float*)d_in[0];   // [B, C, T] fp32
  const float* cb = (const float*)d_in[1];   // [K, V, C] fp32
  float* out = (float*)d_out;

  // workspace carve-up (~98.3 MB)
  char* ws = (char*)d_ws;
  float*          R    = (float*)ws;                                   // 64 MiB
  unsigned short* Rh   = (unsigned short*)(ws + ((size_t)64 << 20));   // 32 MiB
  unsigned short* Cbh  = (unsigned short*)(ws + ((size_t)96 << 20));   // 2 MiB
  float*          c2   = (float*)(ws + ((size_t)98 << 20));            // 16 KiB
  int*            idxb = (int*)(ws + ((size_t)98 << 20) + (16 << 10)); // 256 KiB
  float*          loss = (float*)(ws + ((size_t)98 << 20) + (16 << 10) + (256 << 10));

  float* outIdx = out + (size_t)B_ * C_ * T_;

  rvq_prep_cb<<<K_ * V_, 256, 0, stream>>>(cb, Cbh, c2, loss);
  rvq_init<<<NTOK, 256, 0, stream>>>(z, R, Rh);   // 16.7M elems / 256 = 65536 blocks

  for (int lvl = 0; lvl < K_; ++lvl) {
    rvq_argmin<<<NTOK / (8 * 32), 256, 0, stream>>>(
        Rh, Cbh + (size_t)lvl * V_ * C_, c2 + lvl * V_, idxb, outIdx, lvl);
    rvq_update<<<NTOK, 256, 0, stream>>>(R, Rh, cb + (size_t)lvl * V_ * C_, idxb, loss);
  }

  rvq_final<<<NTOK, 256, 0, stream>>>(z, R, out, loss);
}
